// VectorQuantizer_77275051589669
// MI455X (gfx1250) — matrix operand order repaired and checked by reading, not compiled, not run on hardware
//
#include <hip/hip_runtime.h>

// VQ-VAE vector quantizer, fused for MI455X (gfx1250, wave32, WMMA).
// B=32768 rows, K=4096 codes, D=256. Distances are never materialized:
// each block computes ||e||^2 - 2*x.e via split-bf16 WMMA (hi*hi+hi*lo+lo*hi,
// near-f32 accuracy) with a running per-row argmin, then gathers the winning
// embedding rows and accumulates the loss.

#define BQ   32768
#define KQ   4096
#define DQ   256
#define MT   64          // rows of x per block
#define NWAVE 8

typedef __attribute__((ext_vector_type(16))) __bf16 v16bf;
typedef __attribute__((ext_vector_type(8)))  float  v8f;

static __device__ __forceinline__ unsigned short f2bf(float f) {
  unsigned u = __float_as_uint(f);
  unsigned r = (u + 0x7FFFu + ((u >> 16) & 1u)) >> 16;   // round-to-nearest-even
  return (unsigned short)r;
}
static __device__ __forceinline__ float bf2f(unsigned short h) {
  return __uint_as_float(((unsigned)h) << 16);
}

// ---------------------------------------------------------------------------
// Prep: split embedding into bf16 hi/lo, laid out in a contiguous-k B-fragment order:
//   B (32x16, KxN) bf16 fragment, lane L: N = L%16, khalf = L/16,
//   dword v holds K = khalf*16 + 2v, 2v+1.
// Storage: idx = ((n*8 + ks)*32 + lane)*8 + v   (n = code/16, ks = d/32)
// => each wave's B fragment is one contiguous 1KB block -> coalesced b128s.
// ---------------------------------------------------------------------------
__global__ __launch_bounds__(256) void vq_prep_emb(const float* __restrict__ emb,
                                                   unsigned* __restrict__ eHi,
                                                   unsigned* __restrict__ eLo,
                                                   float* __restrict__ lossAcc) {
  int gid = blockIdx.x * 256 + threadIdx.x;        // 0 .. 4096*128-1
  if (gid == 0) *lossAcc = 0.0f;
  int k  = gid >> 7;                               // code
  int d0 = (gid & 127) * 2;                        // even element index
  int ks    = d0 >> 5;
  int kk    = d0 & 31;
  int khalf = kk >> 4;
  int v     = (kk & 15) >> 1;
  int lane  = khalf * 16 + (k & 15);
  int n     = k >> 4;
  size_t idx = ((size_t)(n * 8 + ks) * 32 + lane) * 8 + v;
  float a0 = emb[(size_t)k * DQ + d0];
  float a1 = emb[(size_t)k * DQ + d0 + 1];
  unsigned short h0 = f2bf(a0), h1 = f2bf(a1);
  unsigned short l0 = f2bf(a0 - bf2f(h0)), l1 = f2bf(a1 - bf2f(h1));
  eHi[idx] = (unsigned)h0 | ((unsigned)h1 << 16);
  eLo[idx] = (unsigned)l0 | ((unsigned)l1 << 16);
}

// ---------------------------------------------------------------------------
// ||e_k||^2, one wave32 per code.
// ---------------------------------------------------------------------------
__global__ __launch_bounds__(256) void vq_norms(const float* __restrict__ emb,
                                                float* __restrict__ eNorm) {
  int w = threadIdx.x >> 5, lane = threadIdx.x & 31;
  int k = blockIdx.x * NWAVE + w;
  const float* er = emb + (size_t)k * DQ;
  float s = 0.0f;
  #pragma unroll
  for (int j = 0; j < DQ / 32; ++j) { float v = er[lane + 32 * j]; s += v * v; }
  #pragma unroll
  for (int off = 16; off >= 1; off >>= 1) s += __shfl_xor(s, off, 32);
  if (lane == 0) eNorm[k] = s;
}

// ---------------------------------------------------------------------------
// Main fused kernel: 512 blocks x 256 threads (8 wave32), 64 rows per block.
// ---------------------------------------------------------------------------
__global__ __launch_bounds__(256) void vq_main(const float* __restrict__ x,
                                               const float* __restrict__ emb,
                                               const unsigned* __restrict__ eHi,
                                               const unsigned* __restrict__ eLo,
                                               const float* __restrict__ eNorm,
                                               float* __restrict__ lossAcc,
                                               float* __restrict__ out) {
  // 64KB LDS: [0..8191] x hi A-fragments, [8192..16383] x lo A-fragments.
  // Re-aliased after the WMMA loop for argmin-merge / reduction scratch.
  __shared__ __align__(32) unsigned ldsbuf[16384];
  const int tid  = threadIdx.x;
  const int lane = tid & 31;
  const int w    = tid >> 5;
  const int rowBase = blockIdx.x * MT;

  // ---- Phase 1: convert x tile -> LDS in A-fragment order (hi/lo split) ----
  // A (16x32, MxK) bf16 fragment, contiguous-k order as for B, lane L: M = L%16, khalf = L/16,
//   dword v holds K = khalf*16 + 2v, 2v+1.
  // Storage: idx = ((mt*8 + ks)*32 + lane)*8 + v.
  #pragma unroll 4
  for (int i = 0; i < 32; ++i) {
    int idx = tid + 256 * i;
    int v  = idx & 7;
    int ln = (idx >> 3) & 31;
    int ks = (idx >> 8) & 7;
    int mt = idx >> 11;
    int m     = mt * 16 + (ln & 15);
    int khalf = ln >> 4;
    int kp    = v * 2 + khalf * 16;
    int k     = ks * 32 + kp;
    const float* xr = x + (size_t)(rowBase + m) * DQ + k;
    float a0 = xr[0], a1 = xr[1];
    unsigned short h0 = f2bf(a0), h1 = f2bf(a1);
    unsigned short l0 = f2bf(a0 - bf2f(h0)), l1 = f2bf(a1 - bf2f(h1));
    ldsbuf[idx]        = (unsigned)h0 | ((unsigned)h1 << 16);
    ldsbuf[8192 + idx] = (unsigned)l0 | ((unsigned)l1 << 16);
  }
  __syncthreads();

  // ---- Phase 2: WMMA sweep over all codes with running argmin ----
  const int mt = w & 3;     // row tile (16 rows)
  const int nh = w >> 2;    // code-chunk parity
  float bd[8]; int bi[8];
  #pragma unroll
  for (int r = 0; r < 8; ++r) { bd[r] = 3.0e38f; bi[r] = 0; }

  for (int it = 0; it < (KQ / 16) / 2; ++it) {
    const int c = it * 2 + nh;                   // 16-code chunk
    v8f acc = {};
    #pragma unroll
    for (int ks = 0; ks < 8; ++ks) {
      const int aIdx = ((mt * 8 + ks) * 32 + lane) * 8;
      const size_t bIdx = ((size_t)(c * 8 + ks) * 32 + lane) * 8;
      v16bf ah = *(const v16bf*)&ldsbuf[aIdx];
      v16bf al = *(const v16bf*)&ldsbuf[8192 + aIdx];
      v16bf bh = *(const v16bf*)(eHi + bIdx);
      v16bf bl = *(const v16bf*)(eLo + bIdx);
      // split-bf16 emulated f32 product: hi*hi + hi*lo + lo*hi
      acc = __builtin_amdgcn_wmma_f32_16x16x32_bf16(false, ah, false, bh,
                                                    (short)0, acc, false, false);
      acc = __builtin_amdgcn_wmma_f32_16x16x32_bf16(false, ah, false, bl,
                                                    (short)0, acc, false, false);
      acc = __builtin_amdgcn_wmma_f32_16x16x32_bf16(false, al, false, bh,
                                                    (short)0, acc, false, false);
    }
    if (it + 1 < (KQ / 16) / 2) {               // prefetch next chunk's B frags
      const size_t nb = ((size_t)((it + 1) * 2 + nh) * 8 * 32 + lane) * 8;
      __builtin_prefetch(eHi + nb, 0, 1);
      __builtin_prefetch(eLo + nb, 0, 1);
    }
    // C/D layout: lane holds column N = lane%16; VGPR r holds row M = r + 8*(lane/16).
    const int code = c * 16 + (lane & 15);
    const float nrm = eNorm[code];
    #pragma unroll
    for (int r = 0; r < 8; ++r) {
      float d = nrm - 2.0f * acc[r];            // ||x||^2 constant per row: dropped
      if (d < bd[r] || (d == bd[r] && code < bi[r])) { bd[r] = d; bi[r] = code; }
    }
  }

  // ---- Phase 3: argmin butterfly within each 16-lane half ----
  #pragma unroll
  for (int off = 8; off >= 1; off >>= 1) {
    #pragma unroll
    for (int r = 0; r < 8; ++r) {
      float od = __shfl_xor(bd[r], off, 32);
      int   oi = __shfl_xor(bi[r], off, 32);
      if (od < bd[r] || (od == bd[r] && oi < bi[r])) { bd[r] = od; bi[r] = oi; }
    }
  }
  __syncthreads();                               // fragments dead; re-alias LDS
  float* sd  = (float*)ldsbuf;                   // [2][64]
  int*   si  = (int*)(ldsbuf + 128);             // [2][64]
  int*   qi  = (int*)(ldsbuf + 256);             // [64]
  float* red = (float*)(ldsbuf + 320);           // [256]
  if ((lane & 15) == 0) {                        // lanes 0 (M=r) and 16 (M=r+8)
    int rb = mt * 16 + ((lane >> 4) << 3);
    #pragma unroll
    for (int r = 0; r < 8; ++r) {
      sd[nh * 64 + rb + r] = bd[r];
      si[nh * 64 + rb + r] = bi[r];
    }
  }
  __syncthreads();
  if (tid < 64) {                                // merge the two code partitions
    float d0 = sd[tid], d1 = sd[64 + tid];
    int   i0 = si[tid], i1 = si[64 + tid];
    qi[tid] = (d1 < d0 || (d1 == d0 && i1 < i0)) ? i1 : i0;
  }
  __syncthreads();

  // ---- Phase 4: gather winning rows (L2-resident), write out, loss ----
  float lsum = 0.0f;
  for (int rr = w; rr < MT; rr += NWAVE) {
    const int row  = rowBase + rr;
    const int code = qi[rr];
    const float* er = emb + (size_t)code * DQ;
    const float* xr = x   + (size_t)row  * DQ;
    float*     orow = out + (size_t)row  * DQ;
    #pragma unroll
    for (int j = 0; j < DQ / 32; ++j) {
      const int cidx = lane + 32 * j;
      float q  = er[cidx];
      float xx = xr[cidx];
      float df = xx - q;
      lsum += df * df;
      orow[cidx] = q;                            // quantized_st == quantized numerically
    }
  }
  red[tid] = lsum;
  __syncthreads();
  for (int s = 128; s > 0; s >>= 1) {
    if (tid < s) red[tid] += red[tid + s];
    __syncthreads();
  }
  if (tid == 0) atomicAdd(lossAcc, red[0]);
}

// ---------------------------------------------------------------------------
// vq_loss = (1 + 0.25) * mean((x - q)^2)
// ---------------------------------------------------------------------------
__global__ void vq_final(const float* __restrict__ lossAcc, float* __restrict__ out) {
  if (threadIdx.x == 0 && blockIdx.x == 0)
    out[(size_t)BQ * DQ] = 1.25f * (*lossAcc) / (float)((size_t)BQ * DQ);
}

extern "C" void kernel_launch(void* const* d_in, const int* in_sizes, int n_in,
                              void* d_out, int out_size, void* d_ws, size_t ws_size,
                              hipStream_t stream) {
  const float* x   = (const float*)d_in[0];      // [B, D] f32
  const float* emb = (const float*)d_in[1];      // [K, D] f32
  float* out = (float*)d_out;                    // [B*D] quantized_st ++ [1] vq_loss

  // Workspace layout (bytes): eHi frags 2MB | eLo frags 2MB | norms 16KB | acc 4B
  char* ws = (char*)d_ws;
  unsigned* eHi   = (unsigned*)(ws);
  unsigned* eLo   = (unsigned*)(ws + (size_t)2 * 1024 * 1024);
  float*    eNorm = (float*)   (ws + (size_t)4 * 1024 * 1024);
  float*    acc   = (float*)   (ws + (size_t)4 * 1024 * 1024 + KQ * sizeof(float));

  vq_prep_emb<<<(KQ * (DQ / 2)) / 256, 256, 0, stream>>>(emb, eHi, eLo, acc);
  vq_norms  <<<KQ / NWAVE, 256, 0, stream>>>(emb, eNorm);
  vq_main   <<<BQ / MT, 256, 0, stream>>>(x, emb, eHi, eLo, eNorm, acc, out);
  vq_final  <<<1, 32, 0, stream>>>(acc, out);
}
